// MultiheadAttention_27084063768678
// MI455X (gfx1250) — matrix operand order repaired and checked by reading, not compiled, not run on hardware
//
#include <hip/hip_runtime.h>
#include <cstddef>
#include <cstdint>

// ---------------- CDNA5 / gfx1250 WMMA types ----------------
typedef __attribute__((ext_vector_type(16))) __bf16 v16bf;
typedef __attribute__((ext_vector_type(8)))  float  v8f;
typedef __attribute__((ext_vector_type(4)))  unsigned int v4u;

#define D_MODEL  1024
#define N_HEADS  16
#define HEAD_DIM 64
#define SEQ      2048
#define BATCH    2
#define M_ROWS   (BATCH * SEQ)   // 4096

union F8      { v8f   v; float f[8]; };
union BF16x16 { v16bf v; v4u q[2]; };

__device__ __forceinline__ unsigned short f2bf(float f) {
    unsigned int u = __float_as_uint(f);
    unsigned int r = u + 0x7FFFu + ((u >> 16) & 1u);   // round-to-nearest-even
    return (unsigned short)(r >> 16);
}

// Low 32 bits of a generic pointer into __shared__ == LDS byte offset
// (flat->LDS mapping truncates to addr[31:0], ISA 10.2).
__device__ __forceinline__ unsigned lds_off(const void* p) {
    return (unsigned)(uintptr_t)p;
}

// ---- CDNA5 async copy: global -> LDS, 16B per lane, tracked by ASYNCcnt ----
__device__ __forceinline__ void async_copy_b128(unsigned lds_byte, const void* gsrc) {
    unsigned long long ga = (unsigned long long)(uintptr_t)gsrc;
    asm volatile("global_load_async_to_lds_b128 %0, %1, off"
                 :: "v"(lds_byte), "v"(ga) : "memory");
}
__device__ __forceinline__ void wait_async0() {
    asm volatile("s_wait_asynccnt 0x0" ::: "memory");
}

// 16-bit A-matrix 16x32 fragment (ISA 7.12.2): lane<16 holds M=lane,
// K = k0+{0..7} (halves 0..7) and k0+{16..23} (halves 8..15);
// lane>=16 holds M=lane-16, K = k0+{8..15} and k0+{24..31}.
__device__ __forceinline__ v16bf load_a_frag(const unsigned short* tile, int stride,
                                             int row, int k0, int lane) {
    const unsigned short* p = tile + row * stride + k0 + ((lane & 16) ? 8 : 0);
    BF16x16 f;
    f.q[0] = *reinterpret_cast<const v4u*>(p);
    f.q[1] = *reinterpret_cast<const v4u*>(p + 16);
    return f.v;
}

// 16-bit B-matrix 32x16 fragment from a K-major tile (row stride 64): p points at
// K=k0+8*(lane>>4), N=lane&15; halves 0..7 = K+{0..7}, halves 8..15 = K+{16..23}.
__device__ __forceinline__ v16bf load_b_frag(const unsigned short* p) {
    unsigned short t[16];
#pragma unroll
    for (int s = 0; s < 16; ++s) t[s] = p[(s < 8 ? s : s + 8) * 64];
    BF16x16 f;
    __builtin_memcpy(&f, t, sizeof(t));
    return f.v;
}

// B-fragment (32(K) x 16(N)) from an N-major LDS tile (rows = N, K contiguous
// in each row): lane holds N=n0+(lane&15), K = k0+8*(lane>>4)+{0..7} and +16+{0..7}.
__device__ __forceinline__ v16bf load_b_frag_tr(const unsigned short* tile, int strideH,
                                                int n0, int k0, int lane) {
    const unsigned short* p = tile + (size_t)(n0 + (lane & 15)) * strideH + k0 + ((lane & 16) ? 8 : 0);
    BF16x16 f;
    f.q[0] = *reinterpret_cast<const v4u*>(p);
    f.q[1] = *reinterpret_cast<const v4u*>(p + 16);
    return f.v;
}

__device__ __forceinline__ v8f wmma_bf16(v16bf a, v16bf b, v8f c) {
    return __builtin_amdgcn_wmma_f32_16x16x32_bf16(false, a, false, b,
                                                   (short)0, c, false, false);
}

// ---------------- fp32 -> bf16 convert ----------------
__global__ void cvt_f32_bf16_kernel(const float* __restrict__ in,
                                    unsigned short* __restrict__ out, int n) {
    int i = blockIdx.x * blockDim.x + threadIdx.x;
    if (i < n) out[i] = f2bf(in[i]);
}

// ---------------- GEMM: Y[m,n] = sum_k A[m,k]*W[n,k] + bias[n] ----------------
// A: [4096,1024] bf16 row-major. W: [1024,1024] bf16 row-major (nn.Linear [out,in]).
// mode 0: bf16 output, head-split to [B,H,S,hd].  mode 1: fp32 output [M,D].
// 128 thr = 4 waves; block tile 64x64; K-step 32; double-buffered async staging.
__global__ __launch_bounds__(128)
void gemm_proj_kernel(const unsigned short* __restrict__ A,
                      const unsigned short* __restrict__ W,
                      const float* __restrict__ bias,
                      unsigned short* __restrict__ outb,
                      float* __restrict__ outf, int mode) {
    __shared__ __align__(16) unsigned short As[2][64 * 32];  // [m][k]
    __shared__ __align__(16) unsigned short Ws[2][64 * 32];  // [n][k] (row-major)

    const int tid  = threadIdx.x;
    const int wid  = tid >> 5;
    const int lane = tid & 31;
    const int mblk = blockIdx.x * 64;
    const int nblk = blockIdx.y * 64;
    const int rowA = wid * 16 + (lane & 15);

    // each thread stages 2 x 16B chunks of A and of W per k-step (256 chunks each)
    const int r0 = tid >> 2, c0 = tid & 3;          // chunk 0: rows 0..31
    const int r1 = r0 + 32,  c1 = c0;               // chunk 1: rows 32..63

    auto stage = [&](int buf, int k0) {
        async_copy_b128(lds_off(&As[buf][r0 * 32 + c0 * 8]),
                        A + (size_t)(mblk + r0) * D_MODEL + k0 + c0 * 8);
        async_copy_b128(lds_off(&As[buf][r1 * 32 + c1 * 8]),
                        A + (size_t)(mblk + r1) * D_MODEL + k0 + c1 * 8);
        async_copy_b128(lds_off(&Ws[buf][r0 * 32 + c0 * 8]),
                        W + (size_t)(nblk + r0) * D_MODEL + k0 + c0 * 8);
        async_copy_b128(lds_off(&Ws[buf][r1 * 32 + c1 * 8]),
                        W + (size_t)(nblk + r1) * D_MODEL + k0 + c1 * 8);
    };

    F8 acc[4];
#pragma unroll
    for (int nt = 0; nt < 4; ++nt)
#pragma unroll
        for (int r = 0; r < 8; ++r) acc[nt].f[r] = 0.0f;

    stage(0, 0);
    wait_async0();
    __syncthreads();

    for (int kk = 0; kk < D_MODEL / 32; ++kk) {
        const int bi = kk & 1;
        if (kk + 1 < D_MODEL / 32) stage(bi ^ 1, 32 * (kk + 1));  // prefetch next

        v16bf a = load_a_frag(As[bi], 32, rowA, 0, lane);
#pragma unroll
        for (int nt = 0; nt < 4; ++nt) {
            v16bf b = load_b_frag_tr(Ws[bi], 32, nt * 16, 0, lane);
            acc[nt].v = wmma_bf16(a, b, acc[nt].v);
        }

        wait_async0();        // prefetched tile landed in LDS
        __syncthreads();      // everyone done with buffer bi
    }

    // epilogue: bias + store.  C layout: elem r -> row r + 8*(lane>>4), col lane&15.
#pragma unroll
    for (int nt = 0; nt < 4; ++nt) {
#pragma unroll
        for (int r = 0; r < 8; ++r) {
            const int m = mblk + wid * 16 + r + ((lane >> 4) << 3);
            const int n = nblk + nt * 16 + (lane & 15);
            const float v = acc[nt].f[r] + bias[n];
            if (mode == 0) {
                const int b_ = m >> 11, s = m & 2047, hh = n >> 6, dd = n & 63;
                outb[(((size_t)b_ * N_HEADS + hh) * SEQ + s) * HEAD_DIM + dd] = f2bf(v);
            } else {
                outf[(size_t)m * D_MODEL + n] = v;
            }
        }
    }
}

// ---------------- Flash attention ----------------
// One block = one (b, h, 64-query tile). 4 waves, each owns 16 query rows.
// Q/K/V bf16 in [B,H,S,hd]; ctx bf16 out in [B,S,D].
__global__ __launch_bounds__(128)
void flash_attn_kernel(const unsigned short* __restrict__ Q,
                       const unsigned short* __restrict__ K,
                       const unsigned short* __restrict__ V,
                       unsigned short* __restrict__ ctx) {
    __shared__ __align__(16) unsigned short Qs[64 * 64];     // [q][d]
    __shared__ __align__(16) unsigned short Ks[2][64 * 64];  // [key][d] (row-major)
    __shared__ __align__(16) unsigned short Vs[2][64 * 64];  // [key][d]
    __shared__ __align__(16) unsigned short Ps[64 * 64];     // [q][key] probs

    const int tid  = threadIdx.x;
    const int wid  = tid >> 5;
    const int lane = tid & 31;
    const int q0   = blockIdx.x * 64;
    const int h    = blockIdx.y;
    const int b    = blockIdx.z;
    const int m0   = wid * 16;
    const int rowA = m0 + (lane & 15);

    const size_t headBase = ((size_t)b * N_HEADS + h) * SEQ;
    const unsigned short* Qg = Q + (headBase + q0) * HEAD_DIM;

    // 64x64-half tile = 512 x 16B chunks; 4 chunks per thread.
    auto stage_kv = [&](int buf, int kb) {
        const unsigned short* Kg = K + (headBase + kb * 64) * HEAD_DIM;
        const unsigned short* Vg = V + (headBase + kb * 64) * HEAD_DIM;
#pragma unroll
        for (int i = 0; i < 4; ++i) {
            const int idx = tid + 128 * i;
            const int r = idx >> 3, c = idx & 7;
            async_copy_b128(lds_off(&Ks[buf][r * 64 + c * 8]),
                            Kg + (size_t)r * HEAD_DIM + c * 8);
            async_copy_b128(lds_off(&Vs[buf][r * 64 + c * 8]),
                            Vg + (size_t)r * HEAD_DIM + c * 8);
        }
    };

    // stage Q once + first K/V block
#pragma unroll
    for (int i = 0; i < 4; ++i) {
        const int idx = tid + 128 * i;
        const int r = idx >> 3, c = idx & 7;
        async_copy_b128(lds_off(&Qs[r * 64 + c * 8]), Qg + (size_t)r * HEAD_DIM + c * 8);
    }
    stage_kv(0, 0);
    wait_async0();
    __syncthreads();

    float mrun[8], lrun[8];
    F8 acc[4];
#pragma unroll
    for (int r = 0; r < 8; ++r) { mrun[r] = -1e30f; lrun[r] = 0.0f; }
#pragma unroll
    for (int nt = 0; nt < 4; ++nt)
#pragma unroll
        for (int r = 0; r < 8; ++r) acc[nt].f[r] = 0.0f;

    for (int kb = 0; kb < SEQ / 64; ++kb) {
        const int bi = kb & 1;
        if (kb + 1 < SEQ / 64) stage_kv(bi ^ 1, kb + 1);   // async prefetch next K/V

        // ---- scores S = Q K^T (16 queries x 64 keys per wave) ----
        F8 sc[4];
#pragma unroll
        for (int nt = 0; nt < 4; ++nt)
#pragma unroll
            for (int r = 0; r < 8; ++r) sc[nt].f[r] = 0.0f;

#pragma unroll
        for (int t = 0; t < 2; ++t) {
            v16bf a = load_a_frag(Qs, 64, rowA, 32 * t, lane);
#pragma unroll
            for (int nt = 0; nt < 4; ++nt) {
                // B-frag: contraction over d, N = keys -> plain b128 loads from
                // row-major K tile (d contiguous in each key row)
                v16bf bfr = load_b_frag_tr(Ks[bi], 64, nt * 16, 32 * t, lane);
                sc[nt].v = wmma_bf16(a, bfr, sc[nt].v);
            }
        }

        // ---- online softmax (C layout: elem r -> row r+8*(lane>>4), col lane&15) ----
        float bm[8];
#pragma unroll
        for (int r = 0; r < 8; ++r) {
#pragma unroll
            for (int nt = 0; nt < 4; ++nt) sc[nt].f[r] *= 0.125f;  // 1/sqrt(64)
            float m4 = fmaxf(fmaxf(sc[0].f[r], sc[1].f[r]), fmaxf(sc[2].f[r], sc[3].f[r]));
            m4 = fmaxf(m4, __shfl_xor(m4, 1, 32));
            m4 = fmaxf(m4, __shfl_xor(m4, 2, 32));
            m4 = fmaxf(m4, __shfl_xor(m4, 4, 32));
            m4 = fmaxf(m4, __shfl_xor(m4, 8, 32));
            bm[r] = m4;
        }
#pragma unroll
        for (int r = 0; r < 8; ++r) {
            const float mnew  = fmaxf(mrun[r], bm[r]);
            const float alpha = __expf(mrun[r] - mnew);
            mrun[r] = mnew;
            float rs = 0.0f;
#pragma unroll
            for (int nt = 0; nt < 4; ++nt) {
                sc[nt].f[r] = __expf(sc[nt].f[r] - mnew);
                rs += sc[nt].f[r];
            }
            rs += __shfl_xor(rs, 1, 32);
            rs += __shfl_xor(rs, 2, 32);
            rs += __shfl_xor(rs, 4, 32);
            rs += __shfl_xor(rs, 8, 32);
            lrun[r] = lrun[r] * alpha + rs;
#pragma unroll
            for (int nt = 0; nt < 4; ++nt) acc[nt].f[r] *= alpha;
        }

        // ---- write P (own 16-row stripe only; DS in-order within wave) ----
#pragma unroll
        for (int nt = 0; nt < 4; ++nt)
#pragma unroll
            for (int r = 0; r < 8; ++r)
                Ps[(m0 + r + ((lane >> 4) << 3)) * 64 + nt * 16 + (lane & 15)] =
                    f2bf(sc[nt].f[r]);

        // ---- ctx += P V  (V is K-major for this product: strided 16-bit loads, stride 64) ----
#pragma unroll
        for (int t = 0; t < 2; ++t) {
            v16bf a = load_a_frag(Ps, 64, rowA, 32 * t, lane);
#pragma unroll
            for (int nt = 0; nt < 4; ++nt) {
                v16bf bfr = load_b_frag(Vs[bi] + (32 * t + ((lane >> 4) << 3)) * 64 + nt * 16 + (lane & 15));
                acc[nt].v = wmma_bf16(a, bfr, acc[nt].v);
            }
        }

        wait_async0();        // prefetched K/V landed
        __syncthreads();      // all waves done with buffer bi
    }

    // ---- finalize & store ctx as [B,S,D] bf16 ----
#pragma unroll
    for (int nt = 0; nt < 4; ++nt)
#pragma unroll
        for (int r = 0; r < 8; ++r) {
            const int s   = q0 + m0 + r + ((lane >> 4) << 3);
            const int col = h * HEAD_DIM + nt * 16 + (lane & 15);
            ctx[((size_t)b * SEQ + s) * D_MODEL + col] = f2bf(acc[nt].f[r] / lrun[r]);
        }
}

// ---------------- host-side orchestration ----------------
extern "C" void kernel_launch(void* const* d_in, const int* in_sizes, int n_in,
                              void* d_out, int out_size, void* d_ws, size_t ws_size,
                              hipStream_t stream) {
    const float* x  = (const float*)d_in[0];
    const float* Wq = (const float*)d_in[1];
    const float* bq = (const float*)d_in[2];
    const float* Wk = (const float*)d_in[3];
    const float* bk = (const float*)d_in[4];
    const float* Wv = (const float*)d_in[5];
    const float* bv = (const float*)d_in[6];
    const float* Wo = (const float*)d_in[7];
    const float* bo = (const float*)d_in[8];

    char* ws = (char*)d_ws;
    const size_t SZ_X = (size_t)M_ROWS * D_MODEL * sizeof(unsigned short);  // 8 MB
    const size_t SZ_W = (size_t)D_MODEL * D_MODEL * sizeof(unsigned short); // 2 MB
    unsigned short* xb   = (unsigned short*)(ws);
    unsigned short* Wqb  = (unsigned short*)(ws + SZ_X);
    unsigned short* Wkb  = (unsigned short*)(ws + SZ_X + SZ_W);
    unsigned short* Wvb  = (unsigned short*)(ws + SZ_X + 2 * SZ_W);
    unsigned short* Wob  = (unsigned short*)(ws + SZ_X + 3 * SZ_W);
    unsigned short* Qb   = (unsigned short*)(ws + SZ_X + 4 * SZ_W);
    unsigned short* Kb   = (unsigned short*)(ws + 2 * SZ_X + 4 * SZ_W);
    unsigned short* Vb   = (unsigned short*)(ws + 3 * SZ_X + 4 * SZ_W);
    unsigned short* ctxb = (unsigned short*)(ws + 4 * SZ_X + 4 * SZ_W);

    const int nX = M_ROWS * D_MODEL;          // 4,194,304
    const int nW = D_MODEL * D_MODEL;         // 1,048,576
    cvt_f32_bf16_kernel<<<(nX + 255) / 256, 256, 0, stream>>>(x,  xb,  nX);
    cvt_f32_bf16_kernel<<<(nW + 255) / 256, 256, 0, stream>>>(Wq, Wqb, nW);
    cvt_f32_bf16_kernel<<<(nW + 255) / 256, 256, 0, stream>>>(Wk, Wkb, nW);
    cvt_f32_bf16_kernel<<<(nW + 255) / 256, 256, 0, stream>>>(Wv, Wvb, nW);
    cvt_f32_bf16_kernel<<<(nW + 255) / 256, 256, 0, stream>>>(Wo, Wob, nW);

    dim3 gGemm(M_ROWS / 64, D_MODEL / 64);    // 64 x 16
    gemm_proj_kernel<<<gGemm, 128, 0, stream>>>(xb, Wqb, bq, Qb, nullptr, 0);
    gemm_proj_kernel<<<gGemm, 128, 0, stream>>>(xb, Wkb, bk, Kb, nullptr, 0);
    gemm_proj_kernel<<<gGemm, 128, 0, stream>>>(xb, Wvb, bv, Vb, nullptr, 0);

    dim3 gAttn(SEQ / 64, N_HEADS, BATCH);     // 32 x 16 x 2
    flash_attn_kernel<<<gAttn, 128, 0, stream>>>(Qb, Kb, Vb, ctxb);

    gemm_proj_kernel<<<gGemm, 128, 0, stream>>>(ctxb, Wob, bo, nullptr, (float*)d_out, 1);
}
